// MultiHeadCrossAttention_v2_41540923686964
// MI455X (gfx1250) — matrix operand order repaired and checked by reading, not compiled, not run on hardware
//
#include <hip/hip_runtime.h>

// ---------------- types ----------------
typedef __bf16 bf16_t;
typedef __attribute__((ext_vector_type(16))) __bf16 v16bf;
typedef __attribute__((ext_vector_type(8)))  __bf16 v8bf;
typedef __attribute__((ext_vector_type(4)))  __bf16 v4bf;
typedef __attribute__((ext_vector_type(8)))  float  v8f;
typedef __attribute__((ext_vector_type(4)))  int    v4i;

#define DIM   1024
#define BATCH 4
#define SEQ   1024
#define HEADS 16
#define DH    64
#define KTOT  2048   // concat key length

// ---------------- CDNA5 async global->LDS path ----------------
#if defined(__has_builtin)
#if __has_builtin(__builtin_amdgcn_global_load_async_to_lds_b128) && \
    __has_builtin(__builtin_amdgcn_s_wait_asynccnt)
#define HAVE_ASYNC_LDS 1
#endif
#endif
#ifndef HAVE_ASYNC_LDS
#define HAVE_ASYNC_LDS 0
#endif

typedef __attribute__((address_space(1))) v4i* gptr_v4i;
typedef __attribute__((address_space(3))) v4i* lptr_v4i;

__device__ __forceinline__ void cp_async16(const bf16_t* g, bf16_t* l) {
#if HAVE_ASYNC_LDS
  // AS1 flat address == global address; AS3: flat LDS-aperture addresses keep
  // the LDS byte offset in the low 32 bits (ISA 10.2), so truncation is valid.
  gptr_v4i gp = (gptr_v4i)(uintptr_t)g;
  lptr_v4i lp = (lptr_v4i)(unsigned)(uintptr_t)l;
  __builtin_amdgcn_global_load_async_to_lds_b128(gp, lp, 0, 0);
#else
  *(v8bf*)l = *(const v8bf*)g;
#endif
}

__device__ __forceinline__ void cp_async_wait() {
#if HAVE_ASYNC_LDS
  __builtin_amdgcn_s_wait_asynccnt(0);
#endif
}

__device__ __forceinline__ v8f wmma_bf16(v16bf a, v16bf b, v8f c) {
  return __builtin_amdgcn_wmma_f32_16x16x32_bf16(false, a, false, b, (short)0, c,
                                                 false, false);
}

union U16 { v16bf v; v8bf h[2]; };

// A-fragment (16x32, lane = M row in [0,15], kh = lane>>4 selects K half):
// VGPR0-3 hold K = kh*8 + 0..7 ; VGPR4-7 hold K = 16 + kh*8 + 0..7
__device__ __forceinline__ v16bf frag_a(const bf16_t* rp, int kh) {
  U16 u;
  u.h[0] = *(const v8bf*)(rp + kh * 8);
  u.h[1] = *(const v8bf*)(rp + 16 + kh * 8);
  return u.v;
}

// B-fragment (32x16, lane = N col, kh selects K half): K = kh*8 + 0..7, then 16 + kh*8 + 0..7
__device__ __forceinline__ v16bf frag_b(const bf16_t* rp, int kh) {
  U16 u;
  const v8bf* p = (const v8bf*)(rp + kh * 8);
  u.h[0] = p[0];
  u.h[1] = p[2];
  return u.v;
}

// ---------------- fp32 -> bf16 convert ----------------
__global__ void k_cvt(const float* __restrict__ s, bf16_t* __restrict__ d, int n) {
  int i = (blockIdx.x * blockDim.x + threadIdx.x) * 4;
  if (i >= n) return;
  float4 f = *(const float4*)(s + i);
  v4bf o;
  o.x = (bf16_t)f.x; o.y = (bf16_t)f.y; o.z = (bf16_t)f.z; o.w = (bf16_t)f.w;
  *(v4bf*)(d + i) = o;
}

// ---------------- GEMM: C = A[M,1024] @ W[1024,1024]^T + bias ----------------
// MODE 0: bf16 row-major out   MODE 1: bf16 V-transposed out   MODE 2: f32 out
#define MT 128
#define NT 128
#define KT 64
#define LDT 72  // padded LDS row stride (elements) to spread banks
#define NKIT (DIM / KT)

template <int MODE>
__global__ __launch_bounds__(256) void k_gemm(const bf16_t* __restrict__ A,
                                              const bf16_t* __restrict__ W,
                                              const float* __restrict__ bias,
                                              void* __restrict__ outp, int joff) {
  __shared__ bf16_t sA[2][MT * LDT];
  __shared__ bf16_t sW[2][NT * LDT];

  const int tid  = threadIdx.x;
  const int lane = tid & 31;
  const int wave = tid >> 5;
  const int wm   = wave & 3;   // 4 waves along M (32 rows each)
  const int wn   = wave >> 2;  // 2 waves along N (64 cols each)
  const int m0   = blockIdx.x * MT;
  const int n0   = blockIdx.y * NT;
  const int lr   = lane & 15;
  const int kh   = lane >> 4;

  v8f acc[2][4];
#pragma unroll
  for (int i = 0; i < 2; i++)
#pragma unroll
    for (int j = 0; j < 4; j++) acc[i][j] = (v8f)(0.0f);

  // stage one 128xKT tile of A and of W into LDS buffer `buf` (async if avail)
  auto stage = [&](int buf, int kk) {
#pragma unroll
    for (int c = 0; c < 4; c++) {
      int chunk = tid + c * 256;   // 1024 chunks of 8 bf16 per array
      int row = chunk >> 3;
      int cc  = (chunk & 7) * 8;
      cp_async16(A + (size_t)(m0 + row) * DIM + kk + cc, &sA[buf][row * LDT + cc]);
      cp_async16(W + (size_t)(n0 + row) * DIM + kk + cc, &sW[buf][row * LDT + cc]);
    }
  };

  stage(0, 0);

  for (int it = 0; it < NKIT; ++it) {
    cp_async_wait();      // my tile-it async loads landed in LDS
    __syncthreads();      // everyone's landed; prior reads of other buf retired
    if (it + 1 < NKIT) stage((it + 1) & 1, (it + 1) * KT);  // overlap w/ compute

    const bf16_t* bA = sA[it & 1];
    const bf16_t* bW = sW[it & 1];
#pragma unroll
    for (int ks = 0; ks < 2; ks++) {  // two K=32 WMMA steps per KT=64 tile
      v16bf af[2], bfr[4];
#pragma unroll
      for (int mi = 0; mi < 2; mi++)
        af[mi] = frag_a(bA + (wm * 32 + mi * 16 + lr) * LDT + ks * 32, kh);
#pragma unroll
      for (int ni = 0; ni < 4; ni++)
        bfr[ni] = frag_b(bW + (wn * 64 + ni * 16 + lr) * LDT + ks * 32, kh);
#pragma unroll
      for (int mi = 0; mi < 2; mi++)
#pragma unroll
        for (int ni = 0; ni < 4; ni++)
          acc[mi][ni] = wmma_bf16(af[mi], bfr[ni], acc[mi][ni]);
    }
  }

  // epilogue: C layout is lane = N (0..15), vgpr r = M row (r + 8*kh)
#pragma unroll
  for (int mi = 0; mi < 2; mi++)
#pragma unroll
    for (int ni = 0; ni < 4; ni++) {
      int nG = n0 + wn * 64 + ni * 16 + lr;
      float bv = bias[nG];
#pragma unroll
      for (int r = 0; r < 8; r++) {
        int mG = m0 + wm * 32 + mi * 16 + r + kh * 8;
        float v = acc[mi][ni][r] + bv;
        if (MODE == 2) {
          ((float*)outp)[(size_t)mG * DIM + nG] = v;
        } else if (MODE == 0) {
          ((bf16_t*)outp)[(size_t)mG * DIM + nG] = (bf16_t)v;
        } else {  // V transposed: Vt[b][h][d][joff + s]
          int bb = mG >> 10, ss = mG & (SEQ - 1);
          int hh = nG >> 6, dd = nG & (DH - 1);
          ((bf16_t*)outp)[((size_t)(bb * HEADS + hh) * DH + dd) * KTOT + joff + ss] =
              (bf16_t)v;
        }
      }
    }
}

// ---------------- flash attention ----------------
// grid: (BATCH*HEADS, SEQ/64), block: 128 (4 waves, each owns 16 Q rows)
__global__ __launch_bounds__(128) void k_attn(const bf16_t* __restrict__ Q,
                                              const bf16_t* __restrict__ Kx,
                                              const bf16_t* __restrict__ Ky,
                                              const bf16_t* __restrict__ Vt,
                                              bf16_t* __restrict__ O) {
  __shared__ float  sS[4][16][68];
  __shared__ bf16_t sP[4][16][72];
  __shared__ float  sAlpha[4][16];
  __shared__ float  sRsum[4][16];

  const int tid  = threadIdx.x;
  const int lane = tid & 31;
  const int w    = tid >> 5;
  const int lr   = lane & 15;
  const int kh   = lane >> 4;

  const int bh = blockIdx.x;
  const int b  = bh >> 4;
  const int h  = bh & (HEADS - 1);
  const int q0 = blockIdx.y * 64 + w * 16;

  // Q fragments, row = q0 + lr, d split into two K=32 windows
  v16bf qf[2];
  {
    const bf16_t* qrow = Q + ((size_t)(b * SEQ + q0 + lr)) * DIM + h * DH;
    qf[0] = frag_a(qrow, kh);
    qf[1] = frag_a(qrow + 32, kh);
  }

  v8f oacc[4];
#pragma unroll
  for (int t = 0; t < 4; t++) oacc[t] = (v8f)(0.0f);
  float rmax = -3.0e38f, rsum = 0.0f;
  const float scale = 0.125f;  // 1/sqrt(64)

  for (int j0 = 0; j0 < KTOT; j0 += 64) {
    // ---- S = Q @ K^T (64-key tile) ----
#pragma unroll
    for (int t = 0; t < 4; t++) {
      int key = j0 + t * 16 + lr;
      const bf16_t* ksrc = (key < SEQ) ? Kx : Ky;
      const bf16_t* krow =
          ksrc + ((size_t)(b * SEQ + (key & (SEQ - 1)))) * DIM + h * DH;
      v8f s = (v8f)(0.0f);
      s = wmma_bf16(qf[0], frag_b(krow, kh), s);
      s = wmma_bf16(qf[1], frag_b(krow + 32, kh), s);
#pragma unroll
      for (int r = 0; r < 8; r++) sS[w][r + kh * 8][t * 16 + lr] = s[r];
    }
    __syncthreads();

    // ---- online softmax (lanes 0..15, one row each) ----
    if (lane < 16) {
      float mx = rmax;
      for (int j = 0; j < 64; j++) mx = fmaxf(mx, sS[w][lane][j] * scale);
      float alpha = __expf(rmax - mx);
      float sum = rsum * alpha;
      for (int j = 0; j < 64; j++) {
        float p = __expf(sS[w][lane][j] * scale - mx);
        sum += p;
        sP[w][lane][j] = (bf16_t)p;
      }
      rmax = mx;
      rsum = sum;
      sAlpha[w][lane] = alpha;
    }
    __syncthreads();

    // ---- rescale O, then O += P @ V ----
#pragma unroll
    for (int r = 0; r < 8; r++) {
      float al = sAlpha[w][r + kh * 8];
#pragma unroll
      for (int t = 0; t < 4; t++) oacc[t][r] *= al;
    }
#pragma unroll
    for (int ks = 0; ks < 2; ks++) {
      v16bf pf = frag_a(&sP[w][lr][0] + ks * 32, kh);
#pragma unroll
      for (int t = 0; t < 4; t++) {
        const bf16_t* vrow =
            Vt + ((size_t)(bh * DH + t * 16 + lr)) * KTOT + j0 + ks * 32;
        oacc[t] = wmma_bf16(pf, frag_b(vrow, kh), oacc[t]);
      }
    }
  }

  if (lane < 16) sRsum[w][lane] = rsum;
  __syncthreads();

#pragma unroll
  for (int r = 0; r < 8; r++) {
    float inv = 1.0f / sRsum[w][r + kh * 8];
    int qG = q0 + r + kh * 8;
#pragma unroll
    for (int t = 0; t < 4; t++) {
      int dG = t * 16 + lr;
      O[((size_t)(b * SEQ + qG)) * DIM + h * DH + dG] = (bf16_t)(oacc[t][r] * inv);
    }
  }
}

// ---------------- host orchestration ----------------
extern "C" void kernel_launch(void* const* d_in, const int* in_sizes, int n_in,
                              void* d_out, int out_size, void* d_ws, size_t ws_size,
                              hipStream_t stream) {
  (void)in_sizes; (void)n_in; (void)out_size; (void)ws_size;
  const float* x     = (const float*)d_in[0];
  const float* y     = (const float*)d_in[1];
  const float* W_Kx  = (const float*)d_in[2];
  const float* b_Kx  = (const float*)d_in[3];
  const float* W_Qx  = (const float*)d_in[4];
  const float* b_Qx  = (const float*)d_in[5];
  const float* W_Vx  = (const float*)d_in[6];
  const float* b_Vx  = (const float*)d_in[7];
  const float* W_Ky  = (const float*)d_in[8];
  const float* b_Ky  = (const float*)d_in[9];
  const float* W_Vy  = (const float*)d_in[10];
  const float* b_Vy  = (const float*)d_in[11];
  const float* W_out = (const float*)d_in[12];
  const float* b_out = (const float*)d_in[13];

  // workspace layout (bytes)
  char* ws = (char*)d_ws;
  const size_t SZ_ACT = (size_t)BATCH * SEQ * DIM * sizeof(bf16_t);  // 8 MB
  const size_t SZ_W   = (size_t)DIM * DIM * sizeof(bf16_t);          // 2 MB
  const size_t SZ_VT  = (size_t)BATCH * HEADS * DH * KTOT * sizeof(bf16_t);  // 16 MB
  size_t off = 0;
  bf16_t* xb  = (bf16_t*)(ws + off); off += SZ_ACT;
  bf16_t* yb  = (bf16_t*)(ws + off); off += SZ_ACT;
  bf16_t* wKx = (bf16_t*)(ws + off); off += SZ_W;
  bf16_t* wQx = (bf16_t*)(ws + off); off += SZ_W;
  bf16_t* wVx = (bf16_t*)(ws + off); off += SZ_W;
  bf16_t* wKy = (bf16_t*)(ws + off); off += SZ_W;
  bf16_t* wVy = (bf16_t*)(ws + off); off += SZ_W;
  bf16_t* wOut= (bf16_t*)(ws + off); off += SZ_W;
  bf16_t* Qb  = (bf16_t*)(ws + off); off += SZ_ACT;
  bf16_t* Kxb = (bf16_t*)(ws + off); off += SZ_ACT;
  bf16_t* Kyb = (bf16_t*)(ws + off); off += SZ_ACT;
  bf16_t* Vt  = (bf16_t*)(ws + off); off += SZ_VT;
  bf16_t* ob  = (bf16_t*)(ws + off); off += SZ_ACT;

  const int nAct = BATCH * SEQ * DIM;  // 4M
  const int nW   = DIM * DIM;          // 1M
  k_cvt<<<nAct / 4 / 256, 256, 0, stream>>>(x, xb, nAct);
  k_cvt<<<nAct / 4 / 256, 256, 0, stream>>>(y, yb, nAct);
  k_cvt<<<nW / 4 / 256, 256, 0, stream>>>(W_Kx, wKx, nW);
  k_cvt<<<nW / 4 / 256, 256, 0, stream>>>(W_Qx, wQx, nW);
  k_cvt<<<nW / 4 / 256, 256, 0, stream>>>(W_Vx, wVx, nW);
  k_cvt<<<nW / 4 / 256, 256, 0, stream>>>(W_Ky, wKy, nW);
  k_cvt<<<nW / 4 / 256, 256, 0, stream>>>(W_Vy, wVy, nW);
  k_cvt<<<nW / 4 / 256, 256, 0, stream>>>(W_out, wOut, nW);

  dim3 gg(BATCH * SEQ / MT, DIM / NT);  // (32, 8)
  k_gemm<0><<<gg, 256, 0, stream>>>(xb, wQx, b_Qx, (void*)Qb, 0);
  k_gemm<0><<<gg, 256, 0, stream>>>(xb, wKx, b_Kx, (void*)Kxb, 0);
  k_gemm<0><<<gg, 256, 0, stream>>>(yb, wKy, b_Ky, (void*)Kyb, 0);
  k_gemm<1><<<gg, 256, 0, stream>>>(xb, wVx, b_Vx, (void*)Vt, 0);
  k_gemm<1><<<gg, 256, 0, stream>>>(yb, wVy, b_Vy, (void*)Vt, SEQ);

  k_attn<<<dim3(BATCH * HEADS, SEQ / 64), 128, 0, stream>>>(Qb, Kxb, Kyb, Vt, ob);

  k_gemm<2><<<gg, 256, 0, stream>>>(ob, wOut, b_out, d_out, 0);
}
